// ScaledDotProductAttention_80470507258061
// MI455X (gfx1250) — matrix operand order repaired and checked by reading, not compiled, not run on hardware
//
#include <hip/hip_runtime.h>

// ---------------------------------------------------------------------------
// Scaled dot-product attention (fp32 ref) on MI455X via bf16 WMMA.
//   queries = inputs @ Wq ; keys = values = memory @ Wk
//   logits  = (Q K^T) / 16, length-masked -> softmax -> alignments
//   contexts = alignments @ keys
// d_out = [contexts (B*Q*AD f32)] ++ [alignments (B*Q*M f32)]
// ---------------------------------------------------------------------------

typedef __attribute__((ext_vector_type(16))) __bf16        v16bf;
typedef __attribute__((ext_vector_type(8)))  float         v8f;
typedef __attribute__((ext_vector_type(4)))  float         f32x4;
typedef __attribute__((ext_vector_type(4)))  unsigned int  u32x4;
typedef __attribute__((ext_vector_type(2)))  unsigned int  u32x2;

#define NB     8
#define QL     2048
#define ML     2048
#define DD     512
#define AD     256
#define NEG_INF_F (-4294967295.0f)   // -2^32 + 1
#define LSTRIDE 2052                 // f32 logits stripe stride (pad 4)
#define PSTRIDE 2056                 // bf16 prob stripe stride (pad 8)
#define WSTRIDE 520                  // bf16 LDS weight row stride (pad 8)

// sched_group_barrier masks
#define SGB_WMMA 0x008
#define SGB_VMR  0x020
#define SGB_DSR  0x100
#define SGB_DSW  0x200

union Frag { v16bf v; __bf16 e[16]; unsigned short h[16]; u32x4 q[2]; };

__device__ __forceinline__ unsigned short bfbits(float x) {
  __bf16 b = (__bf16)x;              // hardware cvt, RNE
  return __builtin_bit_cast(unsigned short, b);
}

__device__ __forceinline__ v8f wmma_bf16(const Frag& a, const Frag& b, v8f c) {
  return __builtin_amdgcn_wmma_f32_16x16x32_bf16(false, a.v, false, b.v,
                                                 (short)0, c, false, false);
}

// two 8-bf16 chunks (table k order: 8h..8h+7, 16+8h..) -> one B fragment
__device__ __forceinline__ void load_bfrag(Frag& d, const __bf16* row, int cB) {
  d.q[0] = *(const u32x4*)(row + cB - ((cB & 16) >> 1));
  d.q[1] = *(const u32x4*)(row + cB - ((cB & 16) >> 1) + 16);
}

// ---------------------------------------------------------------------------
// Kernel 0: Wq/Wk (f32, [D][AD] row-major) -> bf16 transposed [AD][D]
// ---------------------------------------------------------------------------
__global__ void wprep_kernel(const float* __restrict__ Wq,
                             const float* __restrict__ Wk,
                             __bf16* __restrict__ WqT,
                             __bf16* __restrict__ WkT) {
  int idx = blockIdx.x * blockDim.x + threadIdx.x;   // 0 .. 2*AD*DD-1
  int sel = idx >> 17;                               // AD*DD = 131072 = 2^17
  int j   = idx & 131071;
  int a = j >> 9, d = j & 511;
  if (sel == 0) WqT[j] = (__bf16)Wq[d * AD + a];
  else          WkT[j] = (__bf16)Wk[d * AD + a];
}

// ---------------------------------------------------------------------------
// Kernel 1: projections.  [16384,512]f32 @ [512,256]bf16 -> bf16 [16384,256]
// grid.y==0: inputs->Qp ; grid.y==1: memory->Kp (+ transposed KpT [B][AD][M])
// Whole 256x512 bf16 weight matrix staged in LDS (266 KB, padded rows) once
// per block; B fragments ds-fed, order pinned so DS reads run ahead of WMMA.
// ---------------------------------------------------------------------------
__global__ void __launch_bounds__(256)
proj_kernel(const float* __restrict__ inputs, const float* __restrict__ memory,
            const __bf16* __restrict__ WqT, const __bf16* __restrict__ WkT,
            __bf16* __restrict__ Qp, __bf16* __restrict__ Kp,
            __bf16* __restrict__ KpT) {
  extern __shared__ char pshm[];
  __bf16* wlds = (__bf16*)pshm;               // [256][WSTRIDE]

  const int src = blockIdx.y;
  const float* X = src ? memory : inputs;
  const __bf16* WT = src ? WkT : WqT;
  __bf16* Orow = src ? Kp : Qp;

  const int lane = threadIdx.x & 31;
  const int wid  = threadIdx.x >> 5;
  const int rowTile = blockIdx.x * 8 + wid;   // 1024 row tiles
  const int hv = lane >> 4;
  const int ln = lane & 15;

  // Stage WT (row-major [256][512]) into padded LDS; one row per thread.
  {
    const int a = threadIdx.x;                // 256 threads, 256 rows
    const u32x4* s = (const u32x4*)(WT + (size_t)a * DD);
    u32x4* d = (u32x4*)(wlds + a * WSTRIDE);
    for (int i = 0; i < 64; ++i) d[i] = s[i]; // 512 bf16 = 64 x 16B
  }

  const float* xrow = X + (size_t)(rowTile * 16 + ln) * DD;

  // Hoist + convert A fragments for all 16 K-steps (read A exactly once).
  Frag afr[16];
#pragma unroll
  for (int k = 0; k < 16; ++k) {
    const int cA = k * 32 + hv * 8;
    f32x4 p0 = *(const f32x4*)(xrow + cA);
    f32x4 p1 = *(const f32x4*)(xrow + cA + 4);
    f32x4 p2 = *(const f32x4*)(xrow + cA + 16);
    f32x4 p3 = *(const f32x4*)(xrow + cA + 20);
#pragma unroll
    for (int i = 0; i < 4; ++i) {
      afr[k].e[i]      = (__bf16)p0[i];
      afr[k].e[4 + i]  = (__bf16)p1[i];
      afr[k].e[8 + i]  = (__bf16)p2[i];
      afr[k].e[12 + i] = (__bf16)p3[i];
    }
  }
  __syncthreads();

  for (int ct = 0; ct < 16; ++ct) {
    const __bf16* wrow = wlds + (ct * 16 + ln) * WSTRIDE;
    v8f acc = {0.f, 0.f, 0.f, 0.f, 0.f, 0.f, 0.f, 0.f};
    Frag bb[2][2];
    load_bfrag(bb[0][0], wrow, 0 * 32 + hv * 16);
    load_bfrag(bb[0][1], wrow, 1 * 32 + hv * 16);
#pragma unroll
    for (int g = 0; g < 8; ++g) {             // groups of 2 k-steps
      if (g < 7) {
        load_bfrag(bb[(g + 1) & 1][0], wrow, (2 * g + 2) * 32 + hv * 16);
        load_bfrag(bb[(g + 1) & 1][1], wrow, (2 * g + 3) * 32 + hv * 16);
      }
      acc = wmma_bf16(afr[2 * g],     bb[g & 1][0], acc);
      acc = wmma_bf16(afr[2 * g + 1], bb[g & 1][1], acc);
    }
    // Pin schedule: [DS x8][W x2][DS x4][W x2]...[W x4] (one group ahead).
    __builtin_amdgcn_sched_group_barrier(SGB_DSR, 8, 0);
#pragma unroll
    for (int t = 0; t < 6; ++t) {
      __builtin_amdgcn_sched_group_barrier(SGB_WMMA, 2, 0);
      __builtin_amdgcn_sched_group_barrier(SGB_DSR, 4, 0);
    }
    __builtin_amdgcn_sched_group_barrier(SGB_WMMA, 4, 0);
#pragma unroll
    for (int r = 0; r < 8; ++r) {
      const int row = rowTile * 16 + r + hv * 8;
      const int col = ct * 16 + ln;
      const __bf16 vv = (__bf16)acc[r];
      Orow[(size_t)row * AD + col] = vv;
      if (src) {
        const int b = row >> 11, m = row & 2047;
        KpT[((size_t)(b * AD) + col) * ML + m] = vv;
      }
    }
  }
}

// ---------------------------------------------------------------------------
// Kernel 2: fused logits + mask + softmax + context GEMM.
// One workgroup (256 thr / 8 waves) per (batch, 16-query tile).
// LDS: 16x2052 f32 logits stripe + 16x2056 bf16 prob stripe (~198 KB).
// ---------------------------------------------------------------------------
__global__ void __launch_bounds__(256)
attn_kernel(const __bf16* __restrict__ Qp, const __bf16* __restrict__ Kp,
            const __bf16* __restrict__ KpT,
            const int* __restrict__ mlens, const int* __restrict__ qlens,
            float* __restrict__ contexts, float* __restrict__ aligns) {
  extern __shared__ float ashm[];
  float*  L     = ashm;                     // [16][LSTRIDE] f32
  float*  red   = ashm + 16 * LSTRIDE;      // [256]
  float*  rmaxs = red + 256;                // [16]
  float*  rsums = rmaxs + 16;               // [16]
  __bf16* Pbf   = (__bf16*)(rsums + 16);    // [16][PSTRIDE] bf16

  const int tid  = threadIdx.x;
  const int lane = tid & 31;
  const int wid  = tid >> 5;
  const int hv   = lane >> 4;
  const int ln   = lane & 15;
  const int b    = blockIdx.y;
  const int q0   = blockIdx.x * 16;

  const int mlen = mlens[b];
  const int qlen = qlens[b];

  // ---- Phase 1: logits stripe (Q K^T)/16, masked, into LDS ----
  Frag afr[8];
  {
    const __bf16* qrow = Qp + ((size_t)(b * QL) + q0 + ln) * AD;
#pragma unroll
    for (int k = 0; k < 8; ++k) {
      const int cA = k * 32 + hv * 8;
      afr[k].q[0] = *(const u32x4*)(qrow + cA);
      afr[k].q[1] = *(const u32x4*)(qrow + cA + 16);
    }
  }
  for (int i = 0; i < 16; ++i) {
    const int nt = wid + i * 8;        // 128 n-tiles split across 8 waves
    const int m0 = nt * 16;
    const __bf16* krow = Kp + ((size_t)(b * ML) + m0 + ln) * AD;
    v8f acc = {0.f, 0.f, 0.f, 0.f, 0.f, 0.f, 0.f, 0.f};
    Frag bb[2][2];
    load_bfrag(bb[0][0], krow, 0 * 32 + hv * 16);
    load_bfrag(bb[0][1], krow, 1 * 32 + hv * 16);
#pragma unroll
    for (int g = 0; g < 4; ++g) {       // groups of 2 k-steps
      if (g < 3) {
        load_bfrag(bb[(g + 1) & 1][0], krow, (2 * g + 2) * 32 + hv * 16);
        load_bfrag(bb[(g + 1) & 1][1], krow, (2 * g + 3) * 32 + hv * 16);
      }
      acc = wmma_bf16(afr[2 * g],     bb[g & 1][0], acc);
      acc = wmma_bf16(afr[2 * g + 1], bb[g & 1][1], acc);
    }
    const int mcol = m0 + ln;
    const bool mok = mcol < mlen;
#pragma unroll
    for (int r = 0; r < 8; ++r) {
      const int qr = r + hv * 8;
      const bool ok = mok && (q0 + qr < qlen);
      L[qr * LSTRIDE + mcol] = ok ? acc[r] * 0.0625f : NEG_INF_F;
    }
    // Warm the wave's next tile (i+1 -> m0+128) in L0/L2.
    if (m0 + 128 < ML) __builtin_prefetch(krow + 128 * AD, 0, 0);
    // Pin: [VM x8][W x2][VM x4][W x2][VM x4][W x4][DSW x8]
    __builtin_amdgcn_sched_group_barrier(SGB_VMR, 8, 0);
    __builtin_amdgcn_sched_group_barrier(SGB_WMMA, 2, 0);
    __builtin_amdgcn_sched_group_barrier(SGB_VMR, 4, 0);
    __builtin_amdgcn_sched_group_barrier(SGB_WMMA, 2, 0);
    __builtin_amdgcn_sched_group_barrier(SGB_VMR, 4, 0);
    __builtin_amdgcn_sched_group_barrier(SGB_WMMA, 4, 0);
    __builtin_amdgcn_sched_group_barrier(SGB_DSW, 8, 0);
  }
  __syncthreads();

  // ---- Phase 2: row softmax; write alignments; bf16 probs into LDS ----
  {
    const int r = tid >> 4;
    const int sub = tid & 15;
    const float* row = L + r * LSTRIDE + sub * 128;
    float mx = -3.4e38f;
    for (int i = 0; i < 128; i += 4) {
      f32x4 v = *(const f32x4*)(row + i);
      mx = fmaxf(fmaxf(fmaxf(mx, v.x), fmaxf(v.y, v.z)), v.w);
    }
    red[tid] = mx;
    __syncthreads();
    if (sub == 0) {
      float m = red[r * 16];
#pragma unroll
      for (int j = 1; j < 16; ++j) m = fmaxf(m, red[r * 16 + j]);
      rmaxs[r] = m;
    }
    __syncthreads();
    const float rm = rmaxs[r];
    float s = 0.f;
    for (int i = 0; i < 128; i += 4) {
      f32x4 v = *(const f32x4*)(row + i);
      s += __expf(v.x - rm) + __expf(v.y - rm) +
           __expf(v.z - rm) + __expf(v.w - rm);
    }
    red[tid] = s;
    __syncthreads();
    if (sub == 0) {
      float t = red[r * 16];
#pragma unroll
      for (int j = 1; j < 16; ++j) t += red[r * 16 + j];
      rsums[r] = t;
    }
    __syncthreads();
    const float inv = 1.0f / rsums[r];
    float*  ga = aligns + ((size_t)(b * QL) + q0 + r) * ML + sub * 128;
    __bf16* pb = Pbf + r * PSTRIDE + sub * 128;
    for (int i = 0; i < 128; i += 4) {
      f32x4 v = *(const f32x4*)(row + i);
      f32x4 p = {__expf(v.x - rm) * inv, __expf(v.y - rm) * inv,
                 __expf(v.z - rm) * inv, __expf(v.w - rm) * inv};
      *(f32x4*)(ga + i) = p;
      u32x2 pk;
      pk.x = (unsigned)bfbits(p.x) | ((unsigned)bfbits(p.y) << 16);
      pk.y = (unsigned)bfbits(p.z) | ((unsigned)bfbits(p.w) << 16);
      *(u32x2*)(pb + i) = pk;
    }
  }
  __syncthreads();

  // ---- Phase 3: contexts = P @ V (V == keys). ds A + double-buffered B ----
  {
    const int a0 = wid * 32;
    const __bf16* prow = Pbf + ln * PSTRIDE;
    const __bf16* vrow0 = KpT + ((size_t)(b * AD) + a0 + ln) * ML;
    const __bf16* vrow1 = vrow0 + (size_t)16 * ML;
    v8f acc0 = {0.f, 0.f, 0.f, 0.f, 0.f, 0.f, 0.f, 0.f};
    v8f acc1 = {0.f, 0.f, 0.f, 0.f, 0.f, 0.f, 0.f, 0.f};
    Frag bb0[2], bb1[2];
    load_bfrag(bb0[0], vrow0, hv * 16);
    load_bfrag(bb1[0], vrow1, hv * 16);
#pragma unroll 2
    for (int kk = 0; kk < 64; ++kk) {
      const int p = kk & 1;
      if (kk < 63) {
        const int cBn = (kk + 1) * 32 + hv * 16;
        load_bfrag(bb0[p ^ 1], vrow0, cBn);
        load_bfrag(bb1[p ^ 1], vrow1, cBn);
      }
      Frag af;
      const int cA = kk * 32 + hv * 8;
      af.q[0] = *(const u32x4*)(prow + cA);        // ds_load_b128
      af.q[1] = *(const u32x4*)(prow + cA + 16);   // ds_load_b128
      acc0 = wmma_bf16(af, bb0[p], acc0);
      acc1 = wmma_bf16(af, bb1[p], acc1);
      // Pin per iteration: next-B VMEM reads, then A DS reads, then WMMAs.
      __builtin_amdgcn_sched_group_barrier(SGB_VMR, 4, 0);
      __builtin_amdgcn_sched_group_barrier(SGB_DSR, 2, 0);
      __builtin_amdgcn_sched_group_barrier(SGB_WMMA, 2, 0);
    }
#pragma unroll
    for (int r = 0; r < 8; ++r) {
      const int q = q0 + r + hv * 8;
      float* crow = contexts + ((size_t)(b * QL) + q) * AD;
      crow[a0 + ln]      = acc0[r];
      crow[a0 + 16 + ln] = acc1[r];
    }
  }
}

// ---------------------------------------------------------------------------
extern "C" void kernel_launch(void* const* d_in, const int* in_sizes, int n_in,
                              void* d_out, int out_size, void* d_ws,
                              size_t ws_size, hipStream_t stream) {
  (void)in_sizes; (void)n_in; (void)out_size; (void)ws_size;
  const float* inputs = (const float*)d_in[0];
  const float* memory = (const float*)d_in[1];
  const int*   mlens  = (const int*)d_in[2];
  const int*   qlens  = (const int*)d_in[3];
  const float* Wq     = (const float*)d_in[4];
  const float* Wk     = (const float*)d_in[5];
  // d_in[6] = Wv: unused (reference bug uses Wk for values)

  char* ws = (char*)d_ws;
  __bf16* WqT = (__bf16*)(ws);                        // 256 KiB
  __bf16* WkT = (__bf16*)(ws + 262144);               // 256 KiB
  __bf16* Qp  = (__bf16*)(ws + 524288);               // 8 MiB
  __bf16* Kp  = (__bf16*)(ws + 524288 + 8388608);     // 8 MiB
  __bf16* KpT = (__bf16*)(ws + 524288 + 2 * 8388608); // 8 MiB

  float* contexts = (float*)d_out;
  float* aligns   = contexts + (size_t)NB * QL * AD;

  wprep_kernel<<<dim3(1024), dim3(256), 0, stream>>>(Wq, Wk, WqT, WkT);

  const size_t wsh = (size_t)256 * WSTRIDE * sizeof(unsigned short); // 266 KB
  hipFuncSetAttribute(reinterpret_cast<const void*>(proj_kernel),
                      hipFuncAttributeMaxDynamicSharedMemorySize, (int)wsh);
  proj_kernel<<<dim3(128, 2), dim3(256), wsh, stream>>>(inputs, memory, WqT,
                                                        WkT, Qp, Kp, KpT);

  const size_t ash =
      (size_t)(16 * LSTRIDE + 256 + 16 + 16) * sizeof(float) +
      (size_t)(16 * PSTRIDE) * sizeof(unsigned short);
  hipFuncSetAttribute(reinterpret_cast<const void*>(attn_kernel),
                      hipFuncAttributeMaxDynamicSharedMemorySize, (int)ash);
  attn_kernel<<<dim3(QL / 16, NB), dim3(256), ash, stream>>>(
      Qp, Kp, KpT, mlens, qlens, contexts, aligns);
}
